// MMoE_31602369364426
// MI455X (gfx1250) — compile-verified, matrix operand order checked by reading, not run on hardware
//
#include <hip/hip_runtime.h>
#include <math.h>

// ---------------------------------------------------------------------------
// MMoE on MI455X (gfx1250): channel-last bf16 WMMA conv-as-GEMM, fp32 accum.
//   - x transposed once to [B][L][C] bf16; conv layers chain channel-last.
//   - A operand = x rows (M=l, K=ci contiguous -> 2x global_load_b128/lane).
//   - B operand = BN-folded packed weights [e][k][cblk][co][32] (2x b128/lane).
//   - No LDS staging / no barriers in the GEMM loop; K-loop fully unrolled
//     via templates -> dense v_wmma_f32_16x16x32_bf16 stream.
//   - Gates (M=8) and gated combine are cheap VALU kernels.
// ---------------------------------------------------------------------------

typedef __bf16 bf16_t;
typedef __attribute__((ext_vector_type(16))) __bf16 v16bf;
typedef __attribute__((ext_vector_type(8)))  __bf16 v8bf;
typedef __attribute__((ext_vector_type(8)))  float  v8f;

#define EPS 1e-5f
#define SLOPE 0.01f

static __device__ __forceinline__ v16bf cat16(v8bf lo, v8bf hi) {
    return __builtin_shufflevector(lo, hi, 0,1,2,3,4,5,6,7,8,9,10,11,12,13,14,15);
}

// ---------------- prep: f32 [B][C][L] -> bf16 [B][L][C] ----------------
__global__ void cvt_transpose_kernel(const float* __restrict__ in, bf16_t* __restrict__ out,
                                     int C, int L_, int n) {
    int idx = blockIdx.x * 256 + threadIdx.x;
    if (idx >= n) return;
    int l = idx % L_;
    int c = (idx / L_) % C;
    int b = idx / (L_ * C);
    out[((size_t)b * L_ + l) * C + c] = (bf16_t)in[idx];
}

// ---------------- prep: fold BN, pack weights [e][k][cblk][co][kk32] ----------------
__global__ void pack_weights_kernel(const float* __restrict__ w, const float* __restrict__ b,
                                    const float* __restrict__ gamma, const float* __restrict__ beta,
                                    const float* __restrict__ mean, const float* __restrict__ var,
                                    bf16_t* __restrict__ wp, float* __restrict__ bias,
                                    int CO, int CI, int Ksz, int total) {
    int idx = blockIdx.x * 256 + threadIdx.x;
    if (idx >= total) return;
    int CBLK = CI >> 5;
    int kk = idx & 31;
    int co = (idx >> 5) % CO;
    int t  = idx / (32 * CO);
    int cb = t % CBLK;  t /= CBLK;
    int k  = t % Ksz;
    int e  = t / Ksz;
    int ci = (cb << 5) + kk;
    int ec = e * CO + co;
    float sc = gamma[ec] / sqrtf(var[ec] + EPS);
    wp[idx] = (bf16_t)(w[(((size_t)ec) * CI + ci) * Ksz + k] * sc);
    if (k == 0 && cb == 0 && kk == 0)
        bias[ec] = (b[ec] - mean[ec]) * sc + beta[ec];
}

// ---------------- core: channel-last conv-as-GEMM with bf16 WMMA ----------------
// X: [Ein][B][L][CIN] bf16; Wp: [E][KSZ][CIN/32][256][32] bf16; Bias: [E][256] f32
// Out: [E][B][L][256] (bf16 and/or f32). WG = 256 thr = 8 waves (4 over l, 2 over co);
// WG tile = 64 l x 128 co; wave tile = 16 l x 64 co (4 WMMA accumulators).
template <int CIN, int KSZ, int PAD>
__global__ __launch_bounds__(256) void conv_wmma_cl_kernel(
    const bf16_t* __restrict__ X, const bf16_t* __restrict__ Wp,
    const float* __restrict__ Bias, bf16_t* __restrict__ OutBf, float* __restrict__ OutF,
    int B_, int L_, int inPerExpert) {
    const int CO = 256;
    const int CBLK = CIN / 32;
    int tid  = threadIdx.x;
    int lane = tid & 31, wave = tid >> 5;
    int s = lane >> 4;          // half-wave K-group select
    int m = lane & 15;          // A row / B,C column index
    int wM = wave & 3, wN = wave >> 2;
    int l0  = blockIdx.x * 64;
    int co0 = blockIdx.y * 128;
    int e = blockIdx.z / B_;
    int b = blockIdx.z % B_;
    const bf16_t* xb = X + ((size_t)(inPerExpert ? e : 0) * B_ + b) * (size_t)L_ * CIN;
    const bf16_t* wb = Wp + (size_t)e * KSZ * CBLK * CO * 32;

    v8f acc[4] = {};
    int mrow = l0 + wM * 16 + m;

    #pragma unroll
    for (int k = 0; k < KSZ; ++k) {
        int row = mrow + k - PAD;
        bool inr = (row >= 0) && (row < L_);
        const bf16_t* xr = xb + (size_t)row * CIN;
        #pragma unroll
        for (int cb = 0; cb < CBLK; ++cb) {
            // A fragment: lane holds row `row`, K = cb*32 + {s*8..+7, 16+s*8..+7}
            v16bf a = {};
            if (inr) {
                const bf16_t* xp = xr + cb * 32;
                v8bf alo = *(const v8bf*)(xp + s * 8);
                v8bf ahi = *(const v8bf*)(xp + 16 + s * 8);
                a = cat16(alo, ahi);
            }
            #pragma unroll
            for (int f = 0; f < 4; ++f) {
                int n = co0 + wN * 64 + f * 16 + m;   // B column = output channel
                const bf16_t* wkp = wb + ((size_t)(k * CBLK + cb) * CO + n) * 32;
                v8bf blo = *(const v8bf*)(wkp + s * 8);
                v8bf bhi = *(const v8bf*)(wkp + 16 + s * 8);
                acc[f] = __builtin_amdgcn_wmma_f32_16x16x32_bf16(
                    false, a, false, cat16(blo, bhi), (short)0, acc[f], false, false);
            }
        }
    }
    // Epilogue: C layout — lanes<16: M=v, N=lane; lanes>=16: M=v+8, N=lane-16
    size_t base = ((size_t)e * B_ + b) * (size_t)L_ * CO;
    #pragma unroll
    for (int f = 0; f < 4; ++f) {
        int co = co0 + wN * 64 + f * 16 + m;
        float bs = Bias[e * CO + co];
        #pragma unroll
        for (int v = 0; v < 8; ++v) {
            int lrow = l0 + wM * 16 + v + s * 8;
            float val = acc[f][v] + bs;
            size_t o = base + (size_t)lrow * CO + co;
            if (OutF)  OutF[o]  = val;
            if (OutBf) OutBf[o] = (bf16_t)val;
        }
    }
}

// ---------------- gate: conv(K=3,Cout=8) + BN + LeakyReLU + softmax(E) ----------------
// Reads original x f32 [B][CI][L]; writes g [B][8][L] f32.
__global__ __launch_bounds__(256) void gate_kernel(
    const float* __restrict__ x, const float* __restrict__ w, const float* __restrict__ b,
    const float* __restrict__ gamma, const float* __restrict__ beta,
    const float* __restrict__ mean, const float* __restrict__ var,
    float* __restrict__ g, int CI, int L_) {
    __shared__ float wf[128 * 3 * 8];
    __shared__ float bf_[8];
    int tid = threadIdx.x;
    for (int idx = tid; idx < CI * 3 * 8; idx += 256) {
        int e = idx & 7;
        int rest = idx >> 3;
        int k = rest % 3, ci = rest / 3;
        float sc = gamma[e] / sqrtf(var[e] + EPS);
        wf[idx] = w[(e * CI + ci) * 3 + k] * sc;   // layout [(ci*3+k)*8 + e]
    }
    if (tid < 8) {
        float sc = gamma[tid] / sqrtf(var[tid] + EPS);
        bf_[tid] = (b[tid] - mean[tid]) * sc + beta[tid];
    }
    __syncthreads();
    int gi = blockIdx.x * 256 + tid;
    int bb = gi / L_, l = gi % L_;
    float acc[8];
    #pragma unroll
    for (int e = 0; e < 8; ++e) acc[e] = bf_[e];
    const float* xb = x + (size_t)bb * CI * L_;
    for (int ci = 0; ci < CI; ++ci) {
        const float* xr = xb + (size_t)ci * L_;
        float x0 = (l > 0)      ? xr[l - 1] : 0.f;
        float x1 = xr[l];
        float x2 = (l < L_ - 1) ? xr[l + 1] : 0.f;
        const float* wr = &wf[ci * 24];
        #pragma unroll
        for (int e = 0; e < 8; ++e)
            acc[e] += x0 * wr[e] + x1 * wr[8 + e] + x2 * wr[16 + e];
    }
    float mx = -1e30f;
    #pragma unroll
    for (int e = 0; e < 8; ++e) {
        acc[e] = acc[e] > 0.f ? acc[e] : SLOPE * acc[e];
        mx = fmaxf(mx, acc[e]);
    }
    float sum = 0.f, ex[8];
    #pragma unroll
    for (int e = 0; e < 8; ++e) { ex[e] = __expf(acc[e] - mx); sum += ex[e]; }
    float inv = 1.f / sum;
    #pragma unroll
    for (int e = 0; e < 8; ++e) g[((size_t)bb * 8 + e) * L_ + l] = ex[e] * inv;
}

// ---------------- combine: gated sum + residual + LeakyReLU ----------------
// ex: [E][B][L][C] bf16 (channel-last); r: [B][L][C] f32; out1/out2: [B][C][L] f32.
__global__ __launch_bounds__(256) void combine_kernel(
    const bf16_t* __restrict__ ex, const float* __restrict__ g1, const float* __restrict__ g2,
    const float* __restrict__ r, float* __restrict__ out1, float* __restrict__ out2,
    int B_, int C, int L_, int E_) {
    size_t idx = (size_t)blockIdx.x * 256 + threadIdx.x;   // over (b, l, c), c fastest
    size_t total = (size_t)B_ * C * L_;
    if (idx >= total) return;
    int c = (int)(idx % C);
    int l = (int)((idx / C) % L_);
    int b = (int)(idx / ((size_t)C * L_));
    size_t strideE = (size_t)B_ * L_ * C;
    // term from experts[0,0]*g[0,0] (constant over batch, faithful to source)
    float e0 = (float)ex[(size_t)l * C + c];
    float s1 = e0 * g1[l];
    float s2 = e0 * g2[l];
    size_t boff = ((size_t)b * L_ + l) * C + c;
    #pragma unroll
    for (int e = 1; e < 8; ++e) {
        float v = (float)ex[(size_t)e * strideE + boff];
        size_t go = ((size_t)b * 8 + e) * L_ + l;
        s1 += v * g1[go];
        s2 += v * g2[go];
    }
    float rv = r[boff];
    float o1 = s1 + rv; o1 = o1 > 0.f ? o1 : SLOPE * o1;
    float o2 = s2 + rv; o2 = o2 > 0.f ? o2 : SLOPE * o2;
    size_t oo = ((size_t)b * C + c) * L_ + l;
    out1[oo] = o1;
    out2[oo] = o2;
    (void)E_;
}

// ---------------------------------------------------------------------------
extern "C" void kernel_launch(void* const* d_in, const int* in_sizes, int n_in,
                              void* d_out, int out_size, void* d_ws, size_t ws_size,
                              hipStream_t stream) {
    const int E = 8, B = 16, CIN = 128, L = 2048, K = 3, CO = 256;
    (void)in_sizes; (void)n_in; (void)out_size; (void)ws_size;

    // Input flattening: recursive insertion order of setup_inputs() dict.
    const float* x = (const float*)d_in[0];
    const float *ew[3], *eb[3], *eg[3], *ebt[3], *em[3], *ev[3];
    for (int li = 0; li < 3; ++li) {
        int base = 1 + li * 6;
        ew[li]  = (const float*)d_in[base + 0];
        eb[li]  = (const float*)d_in[base + 1];
        eg[li]  = (const float*)d_in[base + 2];
        ebt[li] = (const float*)d_in[base + 3];
        em[li]  = (const float*)d_in[base + 4];
        ev[li]  = (const float*)d_in[base + 5];
    }
    const float* g1p[6]; const float* g2p[6]; const float* rp[6];
    for (int i = 0; i < 6; ++i) {
        g1p[i] = (const float*)d_in[19 + i];
        g2p[i] = (const float*)d_in[25 + i];
        rp[i]  = (const float*)d_in[31 + i];
    }

    // Workspace carve-out (256B aligned). Total ~320 MB.
    char* ws = (char*)d_ws;
    auto alloc = [&](size_t bytes) -> void* {
        void* p = (void*)ws;
        ws += (bytes + 255) & ~(size_t)255;
        return p;
    };
    bf16_t* xt    = (bf16_t*)alloc((size_t)B * CIN * L * 2);       // [B][L][CIN]
    bf16_t* wp1   = (bf16_t*)alloc((size_t)E * CO * CIN * K * 2);
    bf16_t* wp2   = (bf16_t*)alloc((size_t)E * CO * 256 * K * 2);
    bf16_t* wp3   = (bf16_t*)alloc((size_t)E * CO * 256 * K * 2);
    bf16_t* wpr   = (bf16_t*)alloc((size_t)CO * CIN * 2);
    float*  bias1 = (float*)alloc((size_t)E * CO * 4);
    float*  bias2 = (float*)alloc((size_t)E * CO * 4);
    float*  bias3 = (float*)alloc((size_t)E * CO * 4);
    float*  biasr = (float*)alloc((size_t)CO * 4);
    bf16_t* bufA  = (bf16_t*)alloc((size_t)E * B * CO * L * 2);    // L1 & L3 out
    bf16_t* bufB  = (bf16_t*)alloc((size_t)E * B * CO * L * 2);    // L2 out
    float*  rbuf  = (float*)alloc((size_t)B * CO * L * 4);         // [B][L][CO]
    float*  g1    = (float*)alloc((size_t)B * E * L * 4);
    float*  g2    = (float*)alloc((size_t)B * E * L * 4);

    // --- prep ---
    {
        int n = B * CIN * L;
        cvt_transpose_kernel<<<(n + 255) / 256, 256, 0, stream>>>(x, xt, CIN, L, n);
    }
    {
        int t1 = E * CO * CIN * K;
        pack_weights_kernel<<<(t1 + 255) / 256, 256, 0, stream>>>(
            ew[0], eb[0], eg[0], ebt[0], em[0], ev[0], wp1, bias1, CO, CIN, K, t1);
        int t2 = E * CO * 256 * K;
        pack_weights_kernel<<<(t2 + 255) / 256, 256, 0, stream>>>(
            ew[1], eb[1], eg[1], ebt[1], em[1], ev[1], wp2, bias2, CO, 256, K, t2);
        pack_weights_kernel<<<(t2 + 255) / 256, 256, 0, stream>>>(
            ew[2], eb[2], eg[2], ebt[2], em[2], ev[2], wp3, bias3, CO, 256, K, t2);
        int tr = CO * CIN;  // E=1, K=1
        pack_weights_kernel<<<(tr + 255) / 256, 256, 0, stream>>>(
            rp[0], rp[1], rp[2], rp[3], rp[4], rp[5], wpr, biasr, CO, CIN, 1, tr);
    }

    // --- expert towers: 3 WMMA conv layers, ping-pong bf16, channel-last ---
    dim3 blk(256);
    dim3 grdE(L / 64, CO / 128, E * B);
    conv_wmma_cl_kernel<128, 3, 1><<<grdE, blk, 0, stream>>>(xt,   wp1, bias1, bufA, nullptr, B, L, 0);
    conv_wmma_cl_kernel<256, 3, 1><<<grdE, blk, 0, stream>>>(bufA, wp2, bias2, bufB, nullptr, B, L, 1);
    conv_wmma_cl_kernel<256, 3, 1><<<grdE, blk, 0, stream>>>(bufB, wp3, bias3, bufA, nullptr, B, L, 1);

    // --- residual 1x1 conv (same kernel, K=1, E=1, f32 out) ---
    dim3 grdR(L / 64, CO / 128, B);
    conv_wmma_cl_kernel<128, 1, 0><<<grdR, blk, 0, stream>>>(xt, wpr, biasr, nullptr, rbuf, B, L, 0);

    // --- gates ---
    {
        int n = B * L;
        gate_kernel<<<(n + 255) / 256, 256, 0, stream>>>(
            x, g1p[0], g1p[1], g1p[2], g1p[3], g1p[4], g1p[5], g1, CIN, L);
        gate_kernel<<<(n + 255) / 256, 256, 0, stream>>>(
            x, g2p[0], g2p[1], g2p[2], g2p[3], g2p[4], g2p[5], g2, CIN, L);
    }

    // --- gated combine + residual + LeakyReLU -> (out1, out2) ---
    {
        float* out1 = (float*)d_out;
        float* out2 = out1 + (size_t)B * CO * L;
        size_t total = (size_t)B * CO * L;
        combine_kernel<<<(unsigned)((total + 255) / 256), 256, 0, stream>>>(
            bufA, g1, g2, rbuf, out1, out2, B, CO, L, E);
    }
}
